// SimpleRNN_35605278883923
// MI455X (gfx1250) — compile-verified, matrix operand order checked by reading, not run on hardware
//
#include <hip/hip_runtime.h>
#include <hip/hip_bf16.h>
#include <stdint.h>

// ---------------------------------------------------------------------------
// Problem constants: T=8192, D=512, M1=M2=2048, K=1024
// ---------------------------------------------------------------------------
#define T_SEQ 8192
#define MDIM  2048

typedef __attribute__((ext_vector_type(16))) __bf16 v16bf;
typedef __attribute__((ext_vector_type(2)))  __bf16 v2bf;
typedef __attribute__((ext_vector_type(8)))  float  v8f;
typedef int v4i_vec __attribute__((vector_size(16)));   // matches builtin proto

#if defined(__AMDGCN__) && __has_builtin(__builtin_amdgcn_global_load_async_to_lds_b128)
#define HAVE_ASYNC_LDS 1
#else
#define HAVE_ASYNC_LDS 0
#endif

#if defined(__AMDGCN__) && __has_builtin(__builtin_amdgcn_fdot2_f32_bf16)
#define HAVE_FDOT2 1
#else
#define HAVE_FDOT2 0
#endif

__device__ __forceinline__ unsigned short f2bf(float f) {
  unsigned u = __float_as_uint(f);
  return (unsigned short)((u + 0x7FFFu + ((u >> 16) & 1u)) >> 16);
}

__device__ __forceinline__ void copy_b128_g2l(const unsigned short* g, unsigned short* l) {
#if HAVE_ASYNC_LDS
  __builtin_amdgcn_global_load_async_to_lds_b128(
      (__attribute__((address_space(1))) v4i_vec*)(unsigned short*)g,
      (__attribute__((address_space(3))) v4i_vec*)l, 0, 0);
#else
  *(uint4*)l = *(const uint4*)g;
#endif
}

__device__ __forceinline__ void wait_async_zero() {
#if HAVE_ASYNC_LDS
#if __has_builtin(__builtin_amdgcn_s_wait_asynccnt)
  __builtin_amdgcn_s_wait_asynccnt(0);
#else
  asm volatile("s_wait_asynccnt 0" ::: "memory");
#endif
#endif
}

union Frag {
  v16bf v;
  uint4 q[2];
};

// ---------------------------------------------------------------------------
// One-shot converters (amortized once per launch; removes per-tile cvt VALU
// from the GEMM inner loop and halves its global traffic).
// ---------------------------------------------------------------------------
__global__ __launch_bounds__(256)
void cvt_bf16(const float* __restrict__ src, unsigned short* __restrict__ dst) {
  size_t i = ((size_t)blockIdx.x * 256 + threadIdx.x) * 8;
  float4 a = *(const float4*)(src + i);
  float4 b = *(const float4*)(src + i + 4);
  unsigned short o[8] = {f2bf(a.x), f2bf(a.y), f2bf(a.z), f2bf(a.w),
                         f2bf(b.x), f2bf(b.y), f2bf(b.z), f2bf(b.w)};
  *(uint4*)(dst + i) = *(uint4*)o;
}

// dst[n*Kdim + k] = bf16(src[k*Ndim + n])   (weights -> N-major, K contiguous)
__global__ __launch_bounds__(256)
void cvt_transpose_bf16(const float* __restrict__ src, unsigned short* __restrict__ dst,
                        int Kdim, int Ndim) {
  __shared__ float tile[32][33];
  const int k0 = blockIdx.y * 32, n0 = blockIdx.x * 32;
  const int tx = threadIdx.x & 31, ty = threadIdx.x >> 5;  // 32 x 8
#pragma unroll
  for (int j = 0; j < 32; j += 8)
    tile[ty + j][tx] = src[(size_t)(k0 + ty + j) * Ndim + (n0 + tx)];
  __syncthreads();
#pragma unroll
  for (int j = 0; j < 32; j += 8)
    dst[(size_t)(n0 + ty + j) * Kdim + (k0 + tx)] = f2bf(tile[tx][ty + j]);
}

// ---------------------------------------------------------------------------
// GEMM: C(MxN) = A(MxK) @ Bt(NxK)^T + bias, A/Bt bf16 row-major-in-K, C fp32.
// 128x128 tile, BK=32, 8 waves (2x4), 4x2 WMMA frags/wave. Double-buffered
// LDS tiles filled by async global->LDS copies (ASYNCcnt), computed from LDS.
// ---------------------------------------------------------------------------
#define BM 128
#define BN 128
#define BK 32
#define LDSP 40   // bf16 row pitch (80B): 16B-aligned chunks, conflict-free

__global__ __launch_bounds__(256)
void gemm_bf16_wmma(const unsigned short* __restrict__ A,
                    const unsigned short* __restrict__ Bt,
                    const float* __restrict__ bias, float* __restrict__ C,
                    int M, int N, int K) {
  __shared__ __align__(16) unsigned short As[2][BM * LDSP];
  __shared__ __align__(16) unsigned short Bs[2][BN * LDSP];

  const int tid  = threadIdx.x;
  const int wv   = tid >> 5;
  const int lane = tid & 31;
  const int hsel = lane >> 4;
  const int r    = lane & 15;

  const int bm = blockIdx.y * BM;
  const int bn = blockIdx.x * BN;
  const int wm = (wv >> 2) * 64;
  const int wn = (wv & 3) * 32;

  // Stage one BMxBK (and BNxBK) bf16 tile: 512 chunks of 8 bf16 each, 2/thread.
  auto issue_tile = [&](int buf, int k0) {
#pragma unroll
    for (int i = 0; i < 2; ++i) {
      int c = tid + 256 * i;
      int row = c >> 2, kc = (c & 3) << 3;
      copy_b128_g2l(A + (size_t)(bm + row) * K + (k0 + kc),
                    &As[buf][row * LDSP + kc]);
      copy_b128_g2l(Bt + (size_t)(bn + row) * K + (k0 + kc),
                    &Bs[buf][row * LDSP + kc]);
    }
  };

  v8f acc[4][2] = {};

  int buf = 0;
  issue_tile(0, 0);
  for (int k0 = 0; k0 < K; k0 += BK) {
    wait_async_zero();       // our async copies for tile k0 have landed in LDS
    __syncthreads();         // everyone's copies landed
    if (k0 + BK < K) issue_tile(buf ^ 1, k0 + BK);  // prefetch next tile

    // Fragment loads per documented 16-bit layout:
    // lane<16 holds K{0..7,16..23}; lane>=16 holds K{8..15,24..31}
    Frag a[4], b[2];
#pragma unroll
    for (int mi = 0; mi < 4; ++mi) {
      const uint4* p = (const uint4*)&As[buf][(wm + mi * 16 + r) * LDSP];
      a[mi].q[0] = p[hsel];
      a[mi].q[1] = p[2 + hsel];
    }
#pragma unroll
    for (int ni = 0; ni < 2; ++ni) {
      const uint4* p = (const uint4*)&Bs[buf][(wn + ni * 16 + r) * LDSP];
      b[ni].q[0] = p[hsel];
      b[ni].q[1] = p[2 + hsel];
    }
#pragma unroll
    for (int mi = 0; mi < 4; ++mi)
#pragma unroll
      for (int ni = 0; ni < 2; ++ni)
        acc[mi][ni] = __builtin_amdgcn_wmma_f32_16x16x32_bf16(
            false, a[mi].v, false, b[ni].v, (short)0, acc[mi][ni], false, false);

    __syncthreads();         // done reading buf before it is overwritten
    buf ^= 1;
  }

  // Epilogue: D layout -> row = j + 8*hsel, col = r
#pragma unroll
  for (int mi = 0; mi < 4; ++mi) {
#pragma unroll
    for (int ni = 0; ni < 2; ++ni) {
      int gcol = bn + wn + ni * 16 + r;
      float bv = bias[gcol];
      int grow0 = bm + wm + mi * 16 + 8 * hsel;
#pragma unroll
      for (int j = 0; j < 8; ++j)
        C[(size_t)(grow0 + j) * N + gcol] = acc[mi][ni][j] + bv;
    }
  }
}

// ---------------------------------------------------------------------------
// Persistent RNN scan: H[t] = relu(Xp[t] + H[t-1] @ Wh), H[-1]=h0.
// 64 WGs x 256 thr; WG owns 32 cols, wave owns 256 rows; Wh stationary in
// VGPRs as packed bf16 pairs. h broadcast via LDS; v_dot2_f32_bf16 when
// available. Grid sync: monotonic release/acquire counter (64 WGs > 16-WG
// cluster limit, so cluster barriers cannot span the grid).
// ---------------------------------------------------------------------------
#define RNN_WGS  64
#define RNN_COLS 32

__global__ __launch_bounds__(256)
void rnn_scan(const float* __restrict__ Xp, const float* __restrict__ Wh,
              const float* __restrict__ h0, float* __restrict__ H,
              unsigned short* __restrict__ Hbf, unsigned* __restrict__ ctr) {
#if HAVE_FDOT2
  __shared__ __align__(16) unsigned short h_lds[MDIM];  // packed bf16
#else
  __shared__ __align__(16) float h_lds[MDIM];
#endif
  __shared__ float partials[8][RNN_COLS];

  const int tid  = threadIdx.x;
  const int wv   = tid >> 5;
  const int lane = tid & 31;
  const int col  = blockIdx.x * RNN_COLS + lane;
  const int kb   = wv * 256;

  unsigned wreg[128];
#pragma unroll
  for (int i = 0; i < 128; ++i) {
    float w0 = Wh[(size_t)(kb + 2 * i) * MDIM + col];
    float w1 = Wh[(size_t)(kb + 2 * i + 1) * MDIM + col];
    wreg[i] = (unsigned)f2bf(w0) | ((unsigned)f2bf(w1) << 16);
  }

  for (int t = 0; t < T_SEQ; ++t) {
    const float* src = (t == 0) ? h0 : (H + (size_t)(t - 1) * MDIM);
#if HAVE_FDOT2
    {  // stage h as packed bf16: 8 elems/thread -> one 16B LDS store
      float4 a = *(const float4*)(src + tid * 8);
      float4 b = *(const float4*)(src + tid * 8 + 4);
      unsigned short o[8] = {f2bf(a.x), f2bf(a.y), f2bf(a.z), f2bf(a.w),
                             f2bf(b.x), f2bf(b.y), f2bf(b.z), f2bf(b.w)};
      *(uint4*)&h_lds[tid * 8] = *(uint4*)o;
    }
#else
    ((float4*)h_lds)[tid]       = ((const float4*)src)[tid];
    ((float4*)h_lds)[tid + 256] = ((const float4*)src)[tid + 256];
#endif
    __syncthreads();

    float acc = 0.f;
#if HAVE_FDOT2
    const unsigned* hp = (const unsigned*)&h_lds[kb];
#pragma unroll
    for (int i = 0; i < 128; ++i) {
      acc = __builtin_amdgcn_fdot2_f32_bf16(__builtin_bit_cast(v2bf, hp[i]),
                                            __builtin_bit_cast(v2bf, wreg[i]),
                                            acc, false);
    }
#else
#pragma unroll
    for (int i = 0; i < 128; ++i) {
      unsigned w = wreg[i];
      float w0 = __uint_as_float(w << 16);
      float w1 = __uint_as_float(w & 0xFFFF0000u);
      acc = fmaf(h_lds[kb + 2 * i],     w0, acc);
      acc = fmaf(h_lds[kb + 2 * i + 1], w1, acc);
    }
#endif
    partials[wv][lane] = acc;
    __syncthreads();

    if (wv == 0) {
      float s = 0.f;
#pragma unroll
      for (int w = 0; w < 8; ++w) s += partials[w][lane];
      s += Xp[(size_t)t * MDIM + col];
      s = s > 0.f ? s : 0.f;
      H[(size_t)t * MDIM + col]   = s;          // fp32 for the recurrence
      Hbf[(size_t)t * MDIM + col] = f2bf(s);    // bf16 copy for next GEMM
    }

    if (tid == 0) {
      __hip_atomic_fetch_add(ctr, 1u, __ATOMIC_RELEASE, __HIP_MEMORY_SCOPE_AGENT);
      const unsigned target = (unsigned)(t + 1) * RNN_WGS;
      while (__hip_atomic_load(ctr, __ATOMIC_ACQUIRE, __HIP_MEMORY_SCOPE_AGENT) < target)
        __builtin_amdgcn_s_sleep(2);
    }
    __syncthreads();
  }
}

__global__ void zero_ctr(unsigned* p) { *p = 0u; }

// ---------------------------------------------------------------------------
// ws layout (fp32):  B0 Xp(64MB) | B1 H1(64MB) | B2 H2(64MB)
// then bf16: Xbf(8MB) | W1t(2MB) | W2t(8MB) | Wlt(4MB) | H1bf(32MB) | H2bf(32MB) | ctr
// ---------------------------------------------------------------------------
extern "C" void kernel_launch(void* const* d_in, const int* in_sizes, int n_in,
                              void* d_out, int out_size, void* d_ws, size_t ws_size,
                              hipStream_t stream) {
  (void)in_sizes; (void)n_in; (void)out_size; (void)ws_size;

  const float* X   = (const float*)d_in[0];
  const float* Wx1 = (const float*)d_in[1];
  const float* Wh1 = (const float*)d_in[2];
  const float* bh1 = (const float*)d_in[3];
  const float* h01 = (const float*)d_in[4];
  const float* Wx2 = (const float*)d_in[5];
  const float* Wh2 = (const float*)d_in[6];
  const float* bh2 = (const float*)d_in[7];
  const float* h02 = (const float*)d_in[8];
  const float* Wl  = (const float*)d_in[9];
  const float* bl  = (const float*)d_in[10];
  float* out = (float*)d_out;

  const size_t TM = (size_t)T_SEQ * MDIM;        // 16M
  const size_t TD = (size_t)T_SEQ * 512;         // 4M
  float* B0 = (float*)d_ws;
  float* B1 = B0 + TM;
  float* B2 = B1 + TM;
  unsigned short* Xbf  = (unsigned short*)(B2 + TM);
  unsigned short* W1t  = Xbf + TD;                       // 2048 x 512
  unsigned short* W2t  = W1t + (size_t)2048 * 512;       // 2048 x 2048
  unsigned short* Wlt  = W2t + (size_t)2048 * 2048;      // 1024 x 2048
  unsigned short* H1bf = Wlt + (size_t)1024 * 2048;
  unsigned short* H2bf = H1bf + TM;
  unsigned* ctr = (unsigned*)(H2bf + TM);

  // one-shot bf16 conversions / weight transposes
  cvt_bf16<<<(unsigned)(TD / (8 * 256)), 256, 0, stream>>>(X, Xbf);
  cvt_transpose_bf16<<<dim3(2048 / 32, 512 / 32),  256, 0, stream>>>(Wx1, W1t, 512, 2048);
  cvt_transpose_bf16<<<dim3(2048 / 32, 2048 / 32), 256, 0, stream>>>(Wx2, W2t, 2048, 2048);
  cvt_transpose_bf16<<<dim3(1024 / 32, 2048 / 32), 256, 0, stream>>>(Wl,  Wlt, 2048, 1024);

  // Xp1 = X @ Wx1 + bh1
  gemm_bf16_wmma<<<dim3(2048 / BN, T_SEQ / BM), 256, 0, stream>>>(
      Xbf, W1t, bh1, B0, T_SEQ, 2048, 512);
  zero_ctr<<<1, 1, 0, stream>>>(ctr);
  rnn_scan<<<RNN_WGS, 256, 0, stream>>>(B0, Wh1, h01, B1, H1bf, ctr);

  // Xp2 = H1 @ Wx2 + bh2
  gemm_bf16_wmma<<<dim3(2048 / BN, T_SEQ / BM), 256, 0, stream>>>(
      H1bf, W2t, bh2, B0, T_SEQ, 2048, 2048);
  zero_ctr<<<1, 1, 0, stream>>>(ctr);
  rnn_scan<<<RNN_WGS, 256, 0, stream>>>(B0, Wh2, h02, B2, H2bf, ctr);

  // out = H2 @ Wl + bl
  gemm_bf16_wmma<<<dim3(1024 / BN, T_SEQ / BM), 256, 0, stream>>>(
      H2bf, Wlt, bl, out, T_SEQ, 1024, 2048);
}
